// MLA_32968168964723
// MI455X (gfx1250) — matrix operand order repaired and checked by reading, not compiled, not run on hardware
//
#include <hip/hip_runtime.h>

// ---------------- problem constants ----------------
#define BB   2
#define TT   2048
#define DD   2048
#define HH   16
#define HDD  128
#define LL   64
#define NTOK (BB * TT)          // 4096
#define SCALE_ATT 0.125f        // 1/sqrt(64)

typedef _Float16 half_t;
typedef half_t v16h __attribute__((ext_vector_type(16)));
typedef half_t v8h  __attribute__((ext_vector_type(8)));
typedef float  v8f  __attribute__((ext_vector_type(8)));
typedef unsigned int u32x4 __attribute__((ext_vector_type(4)));
typedef int i32x8 __attribute__((ext_vector_type(8)));
typedef int i32x4 __attribute__((ext_vector_type(4)));

// TDM availability. This toolchain (clang-23 / therock HIP headers) exposes the
// 6-arg builtin: (u32x4 g0, i32x8 g1, i32x4 g2, i32x4 g3, i32x8, i32 cpol).
// The guard also keeps the host-compilation Sema pass on the fallback branch.
#if defined(__has_builtin)
#if __has_builtin(__builtin_amdgcn_tensor_load_to_lds) && \
    __has_builtin(__builtin_amdgcn_s_wait_tensorcnt)
#define USE_TDM 1
#endif
#endif
#ifndef USE_TDM
#define USE_TDM 0
#endif

// Build a 16x32 f16 A-fragment: lane m=l&15, kh=l>>4 holds
// halves[0..15] = K = kh*16..+15 (contiguous, matching B fragments).
// Also used to assemble 16 contiguous halves from two 16B-aligned chunks.
__device__ __forceinline__ v16h load_frag16(const half_t* p0, const half_t* p1) {
  v8h a = *(const v8h*)p0;
  v8h b = *(const v8h*)p1;
  v16h r;
#pragma unroll
  for (int i = 0; i < 8; ++i) { r[i] = a[i]; r[i + 8] = b[i]; }
  return r;
}

// ---------------- fp32 -> f16 converts ----------------
__global__ void cvt_f32_f16(const float* __restrict__ s, half_t* __restrict__ d, long n) {
  long i = (long)blockIdx.x * blockDim.x + threadIdx.x;
  long st = (long)gridDim.x * blockDim.x;
  for (; i < n; i += st) d[i] = (half_t)s[i];
}

// src [K][N] fp32 (row-major)  ->  dst [N][K] f16 (transposed)
__global__ void cvt_t_f32_f16(const float* __restrict__ s, half_t* __restrict__ d,
                              int K, int N) {
  long tot = (long)K * N;
  long i = (long)blockIdx.x * blockDim.x + threadIdx.x;
  long st = (long)gridDim.x * blockDim.x;
  for (; i < tot; i += st) {
    int n = (int)(i % N);
    int k = (int)(i / N);
    d[(long)n * K + k] = (half_t)s[i];
  }
}

// =====================================================================
// Big GEMM with TDM-staged, double-buffered B tiles in LDS.
// C[m][n] = sum_k A[m][k] * Wt[n][k] + bias[n]
// block = 128 threads (4 waves); block tile = 64 rows x 64 cols; K step 32.
// B tile (64 cols x 32 k, f16) is shared by all 4 waves -> staged once via
// the Tensor Data Mover with an 80B padded row pitch (bank-conflict free).
// =====================================================================
#define KC   32      // K elements per staged tile
#define BROW 40      // padded LDS row pitch in halves (64B data + 16B pad)

__device__ __forceinline__ void tdm_issue_btile(const half_t* src, int ldw,
                                                unsigned lds_addr) {
#if USE_TDM
  unsigned long long ga = (unsigned long long)(size_t)src;
  // D# group0: count=1 | lds_addr | global_addr[56:0] | type=2
  u32x4 g0 = { 1u, lds_addr, (unsigned)ga,
               (unsigned)((ga >> 32) & 0x01FFFFFFull) | 0x80000000u };
  // D# group1: data_size=2B, pad_enable, pad_interval=16 DWORDs (enc 3),
  // pad_amount=4 DWORDs (enc 3); tensor_dim0=ldw, tensor_dim1=64,
  // tile_dim0=KC, tile_dim1=64, tensor_dim0_stride=ldw.
  i32x8 g1 = { (int)((1u << 16) | (1u << 20) | (3u << 22) | (3u << 25)),
               (int)(((unsigned)ldw & 0xFFFFu) << 16),
               (int)(64u << 16),
               (int)((unsigned)KC << 16),
               64,
               (int)ldw,
               0, 0 };
  i32x4 z4 = { 0, 0, 0, 0 };
  i32x8 z8 = { 0, 0, 0, 0, 0, 0, 0, 0 };
  __builtin_amdgcn_tensor_load_to_lds(g0, g1, z4, z4, z8, 0);
#else
  (void)src; (void)ldw; (void)lds_addr;
#endif
}

template <typename OutT>
__global__ void gemm_wmma_tdm_kernel(const half_t* __restrict__ A, int lda,
                                     const half_t* __restrict__ Wt, int ldw,
                                     const float* __restrict__ bias,
                                     OutT* __restrict__ C, int ldo, int K) {
  __shared__ __align__(128) half_t Bsm[2][64 * BROW];
  const int tid = threadIdx.x;
  const int lane = tid & 31;
  const int wave = tid >> 5;
  const int ml = lane & 15;
  const int kh = lane >> 4;
  const int row0 = (blockIdx.y * 4 + wave) * 16;
  const int col0 = blockIdx.x * 64;

  const half_t* arow = A + (long)(row0 + ml) * lda;
  const half_t* wbase = Wt + (long)col0 * ldw;   // 64 B rows for this block

  const unsigned lds0 = (unsigned)(unsigned long long)(size_t)&Bsm[0][0];
  const unsigned lds1 = (unsigned)(unsigned long long)(size_t)&Bsm[1][0];

  // ---- preload tile k=0 into buffer 0 ----
#if USE_TDM
  if (wave == 0) {
    tdm_issue_btile(wbase, ldw, lds0);
    __builtin_amdgcn_s_wait_tensorcnt(0);
  }
#else
  { // cooperative fallback: 128 threads x 8 halves
    int r = tid >> 1, c = tid & 1;
    *(v8h*)(&Bsm[0][0] + r * BROW + c * 16) =
        *(const v8h*)(wbase + (long)r * ldw + c * 16);
  }
#endif
  __syncthreads();

  const v8f zf = {0.f, 0.f, 0.f, 0.f, 0.f, 0.f, 0.f, 0.f};
  v8f acc[4] = {zf, zf, zf, zf};

  for (int k = 0; k < K; k += KC) {
    const int cb = (k / KC) & 1;
    const half_t* bbuf = &Bsm[cb][0];
    const bool more = (k + KC) < K;

    // kick off DMA of the next tile into the other buffer
    if (more) {
#if USE_TDM
      if (wave == 0)
        tdm_issue_btile(wbase + (k + KC), ldw, cb ? lds0 : lds1);
#else
      int r = tid >> 1, c = tid & 1;
      *(v8h*)(&Bsm[cb ^ 1][0] + r * BROW + c * 16) =
          *(const v8h*)(wbase + (long)r * ldw + (k + KC) + c * 16);
#endif
    }

    // compute on the current tile
    v16h a = load_frag16(arow + k + kh * 16, arow + k + kh * 16 + 8);
#pragma unroll
    for (int nt = 0; nt < 4; ++nt) {
      const half_t* bp = bbuf + (nt * 16 + ml) * BROW + kh * 16;
      v16h b = load_frag16(bp, bp + 8);     // 16 contiguous halves, 16B aligned
      acc[nt] = __builtin_amdgcn_wmma_f32_16x16x32_f16(
          false, a, false, b, (short)0, acc[nt], false, false);
    }

    if (more) {
#if USE_TDM
      if (wave == 0) __builtin_amdgcn_s_wait_tensorcnt(0);
#endif
      __syncthreads();
    }
  }

#pragma unroll
  for (int nt = 0; nt < 4; ++nt) {
    const int col = col0 + nt * 16 + ml;
    const float bv = bias ? bias[col] : 0.f;
#pragma unroll
    for (int r = 0; r < 8; ++r) {
      const int row = row0 + r + kh * 8;    // C/D layout: lanes16-31 -> M=r+8
      C[(long)row * ldo + col] = (OutT)(acc[nt][r] + bv);
    }
  }
}

// =====================================================================
// Small generic WMMA GEMM (K<=128): direct global B loads, grid.z strided.
// =====================================================================
template <typename OutT>
__global__ void gemm_wmma_kernel(const half_t* __restrict__ A, int lda, long zAStep,
                                 const half_t* __restrict__ Wt, int ldw,
                                 const float* __restrict__ bias,
                                 OutT* __restrict__ C, int ldo, long zOStep,
                                 int K) {
  A += (long)blockIdx.z * zAStep;
  C += (long)blockIdx.z * zOStep;
  const int lane = threadIdx.x & 31;
  const int wave = threadIdx.x >> 5;
  const int ml = lane & 15;
  const int kh = lane >> 4;
  const int row0 = (blockIdx.y * 4 + wave) * 16;
  const int col0 = blockIdx.x * 64;

  const half_t* arow = A + (long)(row0 + ml) * lda;
  const half_t* wrow[4];
#pragma unroll
  for (int nt = 0; nt < 4; ++nt)
    wrow[nt] = Wt + (long)(col0 + nt * 16 + ml) * ldw + kh * 16;

  const v8f zf = {0.f, 0.f, 0.f, 0.f, 0.f, 0.f, 0.f, 0.f};
  v8f acc[4] = {zf, zf, zf, zf};

  for (int k = 0; k < K; k += 32) {
    v16h a = load_frag16(arow + k + kh * 16, arow + k + kh * 16 + 8);
#pragma unroll
    for (int nt = 0; nt < 4; ++nt) {
      v16h b = *(const v16h*)(wrow[nt] + k);
      acc[nt] = __builtin_amdgcn_wmma_f32_16x16x32_f16(
          false, a, false, b, (short)0, acc[nt], false, false);
    }
  }

#pragma unroll
  for (int nt = 0; nt < 4; ++nt) {
    const int col = col0 + nt * 16 + ml;
    const float bv = bias ? bias[col] : 0.f;
#pragma unroll
    for (int r = 0; r < 8; ++r) {
      const int row = row0 + r + kh * 8;
      C[(long)row * ldo + col] = (OutT)(acc[nt][r] + bv);
    }
  }
}

// ---------------- RoPE on Q,K (f16 in place, fp32 math) ----------------
__global__ void rope_kernel(half_t* __restrict__ Q, half_t* __restrict__ Kk,
                            const float* __restrict__ freqs) {
  const long tot = (long)NTOK * HH * (HDD / 2);
  long idx = (long)blockIdx.x * blockDim.x + threadIdx.x;
  const long st = (long)gridDim.x * blockDim.x;
  for (; idx < tot; idx += st) {
    const int i = (int)(idx & 63);
    const int h = (int)((idx >> 6) & 15);
    const int n = (int)(idx >> 10);
    const int t = n & (TT - 1);
    const float th = freqs[t * (HDD / 2) + i];
    const float c = cosf(th), s = sinf(th);
    const long base = (long)n * DD + h * HDD + 2 * i;
    float a = (float)Q[base], b = (float)Q[base + 1];
    Q[base] = (half_t)(a * c - b * s);
    Q[base + 1] = (half_t)(a * s + b * c);
    a = (float)Kk[base]; b = (float)Kk[base + 1];
    Kk[base] = (half_t)(a * c - b * s);
    Kk[base + 1] = (half_t)(a * s + b * c);
  }
}

// v_lat [b,t,h,l] -> vt [b,h,l,t]
__global__ void transpose_v_kernel(const half_t* __restrict__ vlat, half_t* __restrict__ vt) {
  const long tot = (long)NTOK * HH * LL;
  long idx = (long)blockIdx.x * blockDim.x + threadIdx.x;
  const long st = (long)gridDim.x * blockDim.x;
  for (; idx < tot; idx += st) {
    const int l = (int)(idx & 63);
    const int h = (int)((idx >> 6) & 15);
    const int t = (int)((idx >> 10) & (TT - 1));
    const int b = (int)(idx >> 21);
    vt[(((long)(b * HH + h) * LL + l) * TT) + t] = vlat[idx];
  }
}

// ---------------- flash attention over latents (one wave / 16 queries) ----------------
__global__ void flash_kernel(const half_t* __restrict__ qlat,
                             const half_t* __restrict__ klat,
                             const half_t* __restrict__ vt,
                             half_t* __restrict__ ctx) {
  __shared__ __align__(64) half_t Pld[16 * 32];   // P tile staging (D-frag -> A-frag)
  const int lane = threadIdx.x;
  const int ml = lane & 15;
  const int kh = lane >> 4;
  const int qtile = blockIdx.x;
  const int b = blockIdx.y >> 4;
  const int h = blockIdx.y & 15;
  const int qbase = qtile * 16;

  const half_t* qrow = qlat + ((long)((long)b * TT + qbase + ml) * HH + h) * LL;
  const v16h qa0 = load_frag16(qrow + kh * 16, qrow + kh * 16 + 8);        // latent 0..31
  const v16h qa1 = load_frag16(qrow + 32 + kh * 16, qrow + 40 + kh * 16);   // latent 32..63

  const v8f zf = {0.f, 0.f, 0.f, 0.f, 0.f, 0.f, 0.f, 0.f};
  v8f acc[4] = {zf, zf, zf, zf};
  float m_r[8], l_r[8];
#pragma unroll
  for (int r = 0; r < 8; ++r) { m_r[r] = -1e30f; l_r[r] = 0.f; }

  const long krow = (long)HH * LL;                            // stride between key rows
  const half_t* kbh = klat + ((long)b * TT * HH + h) * LL;    // key row 0 for (b,h)
  const half_t* vbh = vt + (long)(b * HH + h) * LL * TT;      // [latent][t]

  for (int jb = 0; jb <= qbase; jb += 32) {
    const half_t* k0 = kbh + (long)(jb + ml) * krow + kh * 16;
    const half_t* k1 = k0 + 16 * krow;
    v16h kb00 = *(const v16h*)(k0);
    v16h kb01 = *(const v16h*)(k0 + 32);
    v16h kb10 = *(const v16h*)(k1);
    v16h kb11 = *(const v16h*)(k1 + 32);
    if (jb + 32 <= qbase) __builtin_prefetch(k0 + 32 * krow, 0, 1);  // global_prefetch_b8

    v8f s0 = zf, s1 = zf;
    s0 = __builtin_amdgcn_wmma_f32_16x16x32_f16(false, qa0, false, kb00, (short)0, s0, false, false);
    s0 = __builtin_amdgcn_wmma_f32_16x16x32_f16(false, qa1, false, kb01, (short)0, s0, false, false);
    s1 = __builtin_amdgcn_wmma_f32_16x16x32_f16(false, qa0, false, kb10, (short)0, s1, false, false);
    s1 = __builtin_amdgcn_wmma_f32_16x16x32_f16(false, qa1, false, kb11, (short)0, s1, false, false);

#pragma unroll
    for (int r = 0; r < 8; ++r) {
      const int qi = qbase + r + kh * 8;
      float v0 = s0[r] * SCALE_ATT; if (jb + ml > qi)      v0 = -1e30f;
      float v1 = s1[r] * SCALE_ATT; if (jb + 16 + ml > qi) v1 = -1e30f;
      float mr = fmaxf(v0, v1);
      mr = fmaxf(mr, __shfl_xor(mr, 1));
      mr = fmaxf(mr, __shfl_xor(mr, 2));
      mr = fmaxf(mr, __shfl_xor(mr, 4));
      mr = fmaxf(mr, __shfl_xor(mr, 8));
      const float mnew = fmaxf(m_r[r], mr);
      const float alpha = __expf(m_r[r] - mnew);
      const float p0 = __expf(v0 - mnew);
      const float p1 = __expf(v1 - mnew);
      float sum = p0 + p1;
      sum += __shfl_xor(sum, 1);
      sum += __shfl_xor(sum, 2);
      sum += __shfl_xor(sum, 4);
      sum += __shfl_xor(sum, 8);
      l_r[r] = l_r[r] * alpha + sum;
      m_r[r] = mnew;
#pragma unroll
      for (int nt = 0; nt < 4; ++nt) acc[nt][r] *= alpha;
      const int prow = r + kh * 8;
      Pld[prow * 32 + ml] = (half_t)p0;
      Pld[prow * 32 + 16 + ml] = (half_t)p1;
    }
    asm volatile("" ::: "memory");   // DS ops are in-order per wave

    const half_t* pr = Pld + ml * 32;
    const v16h pa = load_frag16(pr + kh * 16, pr + kh * 16 + 8);   // P as A-frag (K=32 keys)
#pragma unroll
    for (int nt = 0; nt < 4; ++nt) {
      const half_t* vrow = vbh + (long)(nt * 16 + ml) * TT + jb + kh * 16;
      v16h vb = *(const v16h*)vrow;
      acc[nt] = __builtin_amdgcn_wmma_f32_16x16x32_f16(
          false, pa, false, vb, (short)0, acc[nt], false, false);
    }
  }

#pragma unroll
  for (int r = 0; r < 8; ++r) {
    const int tq = qbase + r + kh * 8;
    const float inv = 1.0f / l_r[r];
    const long base = ((long)((long)b * TT + tq) * HH + h) * LL;
#pragma unroll
    for (int nt = 0; nt < 4; ++nt)
      ctx[base + nt * 16 + ml] = (half_t)(acc[nt][r] * inv);
  }
}

// ---------------- host-side launcher ----------------
extern "C" void kernel_launch(void* const* d_in, const int* in_sizes, int n_in,
                              void* d_out, int out_size, void* d_ws, size_t ws_size,
                              hipStream_t stream) {
  (void)in_sizes; (void)n_in; (void)out_size; (void)ws_size;
  const float* x     = (const float*)d_in[0];
  const float* freqs = (const float*)d_in[1];
  // d_in[2] = mask: causal mask applied analytically, not read
  const float* Wq  = (const float*)d_in[3];  const float* bq  = (const float*)d_in[4];
  const float* Wk  = (const float*)d_in[5];  const float* bk  = (const float*)d_in[6];
  const float* Wv  = (const float*)d_in[7];  const float* bv  = (const float*)d_in[8];
  const float* Wo  = (const float*)d_in[9];  const float* bo  = (const float*)d_in[10];
  const float* Wcq = (const float*)d_in[11]; const float* bcq = (const float*)d_in[12];
  const float* Wck = (const float*)d_in[13]; const float* bck = (const float*)d_in[14];
  const float* Wcv = (const float*)d_in[15]; const float* bcv = (const float*)d_in[16];
  const float* Wd  = (const float*)d_in[17]; const float* bd  = (const float*)d_in[18];
  float* out = (float*)d_out;

  char* ws = (char*)d_ws;
  size_t off = 0;
  auto carve = [&](size_t halves) -> half_t* {
    half_t* p = (half_t*)(ws + off);
    off += (halves * sizeof(half_t) + 255) & ~(size_t)255;
    return p;
  };
  half_t* xh   = carve((size_t)NTOK * DD);
  half_t* Wqt  = carve((size_t)DD * DD);
  half_t* Wkt  = carve((size_t)DD * DD);
  half_t* Wvt  = carve((size_t)DD * DD);
  half_t* Wot  = carve((size_t)DD * DD);
  half_t* Wcqt = carve((size_t)HDD * LL);
  half_t* Wckt = carve((size_t)HDD * LL);
  half_t* Wcvt = carve((size_t)HDD * LL);
  half_t* Wdt  = carve((size_t)LL * HDD);
  half_t* Qh   = carve((size_t)NTOK * DD);
  half_t* Kh   = carve((size_t)NTOK * DD);
  half_t* Vh   = carve((size_t)NTOK * DD);
  half_t* qlat = carve((size_t)NTOK * HH * LL);
  half_t* klat = carve((size_t)NTOK * HH * LL);
  half_t* vlat = carve((size_t)NTOK * HH * LL);
  half_t* vtb  = carve((size_t)NTOK * HH * LL);
  half_t* ctx  = carve((size_t)NTOK * HH * LL);
  half_t* uh   = carve((size_t)NTOK * DD);

  // 1) converts (weights transposed to [N][K] f16)
  cvt_f32_f16<<<2048, 256, 0, stream>>>(x, xh, (long)NTOK * DD);
  cvt_t_f32_f16<<<2048, 256, 0, stream>>>(Wq, Wqt, DD, DD);
  cvt_t_f32_f16<<<2048, 256, 0, stream>>>(Wk, Wkt, DD, DD);
  cvt_t_f32_f16<<<2048, 256, 0, stream>>>(Wv, Wvt, DD, DD);
  cvt_t_f32_f16<<<2048, 256, 0, stream>>>(Wo, Wot, DD, DD);
  cvt_t_f32_f16<<<32, 256, 0, stream>>>(Wcq, Wcqt, HDD, LL);
  cvt_t_f32_f16<<<32, 256, 0, stream>>>(Wck, Wckt, HDD, LL);
  cvt_t_f32_f16<<<32, 256, 0, stream>>>(Wcv, Wcvt, HDD, LL);
  cvt_t_f32_f16<<<32, 256, 0, stream>>>(Wd, Wdt, LL, HDD);

  // 2) QKV projections (TDM-staged B tiles): [4096 x 2048] x [2048 x 2048]
  {
    dim3 g(DD / 64, NTOK / 64, 1);
    gemm_wmma_tdm_kernel<half_t><<<g, 128, 0, stream>>>(xh, DD, Wqt, DD, bq, Qh, DD, DD);
    gemm_wmma_tdm_kernel<half_t><<<g, 128, 0, stream>>>(xh, DD, Wkt, DD, bk, Kh, DD, DD);
    gemm_wmma_tdm_kernel<half_t><<<g, 128, 0, stream>>>(xh, DD, Wvt, DD, bv, Vh, DD, DD);
  }

  // 3) RoPE on Q,K
  rope_kernel<<<4096, 256, 0, stream>>>(Qh, Kh, freqs);

  // 4) per-head latent compression -> [b,t,h,L], grid.z = head
  {
    dim3 g(1, NTOK / 64, HH);
    gemm_wmma_kernel<half_t><<<g, 128, 0, stream>>>(Qh, DD, HDD, Wcqt, HDD, bcq,
                                                    qlat, HH * LL, LL, HDD);
    gemm_wmma_kernel<half_t><<<g, 128, 0, stream>>>(Kh, DD, HDD, Wckt, HDD, bck,
                                                    klat, HH * LL, LL, HDD);
    gemm_wmma_kernel<half_t><<<g, 128, 0, stream>>>(Vh, DD, HDD, Wcvt, HDD, bcv,
                                                    vlat, HH * LL, LL, HDD);
  }

  // 5) V -> [b,h,L,T] for contiguous P*V B-fragments
  transpose_v_kernel<<<4096, 256, 0, stream>>>(vlat, vtb);

  // 6) flash attention (online softmax), one wave per 16 queries
  {
    dim3 g(TT / 16, BB * HH, 1);
    flash_kernel<<<g, 32, 0, stream>>>(qlat, klat, vtb, ctx);
  }

  // 7) decompress: rows = (b,t,h) = 65536, K=64 -> 128; output lands as [n][h*128+k]
  {
    dim3 g(HDD / 64, (NTOK * HH) / 64, 1);
    gemm_wmma_kernel<half_t><<<g, 128, 0, stream>>>(ctx, LL, 0, Wdt, LL, bd,
                                                    uh, HDD, 0, LL);
  }

  // 8) output projection (TDM-staged) -> fp32 d_out
  {
    dim3 g(DD / 64, NTOK / 64, 1);
    gemm_wmma_tdm_kernel<float><<<g, 128, 0, stream>>>(uh, DD, Wot, DD, bo,
                                                       out, DD, DD);
  }
}
